// GraphTransformerLayer_30751965839528
// MI455X (gfx1250) — compile-verified, matrix operand order checked by reading, not run on hardware
//
#include <hip/hip_runtime.h>

#define N_NODES 6144
#define DMODEL 128
#define NHEADS 4
#define DHEAD 32
#define WORDS_PER_ROW (N_NODES / 32) /* 192 mask words per query row */

typedef __attribute__((ext_vector_type(16))) _Float16 v16h;
typedef __attribute__((ext_vector_type(8)))  _Float16 v8h;
typedef __attribute__((ext_vector_type(8)))  float    v8f;

static __device__ __forceinline__ v8f wmma_f16(v16h a, v16h b, v8f c) {
  // D = A(16x32 f16) * B(32x16 f16) + C(16x16 f32)
  return __builtin_amdgcn_wmma_f32_16x16x32_f16(false, a, false, b, (short)0, c,
                                                false, false);
}

// Load a 16x32 f16 fragment in the ISA "16-bit A-matrix 16x32" layout.
// lane m = lane&15, lane-group koff = (lane>>4)*8:
//   halves 0..7  -> K = koff+0..7
//   halves 8..15 -> K = 16+koff+0..7
// B-fragments of Y(32x16) load with the same code from Y^T (row-major).
static __device__ __forceinline__ v16h load_frag(const _Float16* __restrict__ base,
                                                 int row0, int col0, int ld, int lane) {
  const int m = lane & 15;
  const int koff = (lane >> 4) << 3;
  const _Float16* p = base + (size_t)(row0 + m) * (size_t)ld + col0;
  v8h lo = *(const v8h*)(p + koff);
  v8h hi = *(const v8h*)(p + 16 + koff);
  v16h a;
#pragma unroll
  for (int j = 0; j < 8; ++j) { a[j] = lo[j]; a[j + 8] = hi[j]; }
  return a;
}

// ---------------- prep kernels ----------------

__global__ void zero_mask_kernel(unsigned* __restrict__ m, int nwords) {
  for (int i = blockIdx.x * blockDim.x + threadIdx.x; i < nwords;
       i += gridDim.x * blockDim.x)
    m[i] = 0u;
}

__global__ void build_mask_kernel(const int* __restrict__ src,
                                  const int* __restrict__ dst,
                                  unsigned* __restrict__ m, int E) {
  int e = blockIdx.x * blockDim.x + threadIdx.x;
  if (e >= E) return;
  int r = src[e];
  int c = dst[e];
  atomicOr(&m[(size_t)r * WORDS_PER_ROW + (c >> 5)], 1u << (c & 31));
}

__global__ void cvt_f16_kernel(const float* __restrict__ x,
                               _Float16* __restrict__ y, int n) {
  for (int i = blockIdx.x * blockDim.x + threadIdx.x; i < n;
       i += gridDim.x * blockDim.x)
    y[i] = (_Float16)x[i];
}

// Wt[o][i] = W[i][o] in f16 (so B-fragments load like A-fragments from Wt).
__global__ void transpose_w_kernel(const float* __restrict__ W,
                                   _Float16* __restrict__ Wt) {
  int t = blockIdx.x * blockDim.x + threadIdx.x; // 0..16383
  int o = t >> 7, i = t & 127;
  Wt[(size_t)o * DMODEL + i] = (_Float16)W[(size_t)i * DMODEL + o];
}

// ---------------- QKV projection (one wave per 16x16 output tile) ----------------

__global__ __launch_bounds__(32)
void qkv_kernel(const _Float16* __restrict__ h16,
                const _Float16* __restrict__ WqT, const _Float16* __restrict__ WkT,
                const _Float16* __restrict__ WvT,
                const float* __restrict__ bq, const float* __restrict__ bk,
                const float* __restrict__ bv,
                _Float16* __restrict__ q16, _Float16* __restrict__ k16,
                _Float16* __restrict__ vT) {
  const int ct = blockIdx.x;    // output col tile (0..7)
  const int rt = blockIdx.y;    // node row tile (0..383)
  const int which = blockIdx.z; // 0=q 1=k 2=v
  const int lane = threadIdx.x;
  const int g = lane >> 4, n = lane & 15;

  const _Float16* Wt = (which == 0) ? WqT : (which == 1) ? WkT : WvT;
  const float* bias = (which == 0) ? bq : (which == 1) ? bk : bv;

  v8f c = {};
#pragma unroll
  for (int kk = 0; kk < 4; ++kk) {
    v16h a = load_frag(h16, rt * 16, kk * 32, DMODEL, lane);
    v16h b = load_frag(Wt, ct * 16, kk * 32, DMODEL, lane);
    c = wmma_f16(a, b, c);
  }
  const int col = ct * 16 + n;
  const float bcol = bias[col];
  if (which < 2) {
    // q is pre-scaled by 1/sqrt(DH) so attention logits need no extra scale
    const float scale = (which == 0) ? 0.17677669529663687f : 1.0f;
    _Float16* dst = (which == 0) ? q16 : k16;
#pragma unroll
    for (int r = 0; r < 8; ++r) {
      const int row = rt * 16 + r + g * 8;
      dst[(size_t)row * DMODEL + col] = (_Float16)((c[r] + bcol) * scale);
    }
  } else {
    // store V transposed: vT[d (0..127)][node]; contiguous 8 nodes per lane
    v8h pack;
#pragma unroll
    for (int r = 0; r < 8; ++r) pack[r] = (_Float16)(c[r] + bcol);
    *(v8h*)(vT + (size_t)col * N_NODES + rt * 16 + g * 8) = pack;
  }
}

// ---------------- flash masked attention (one wave per 16-query tile per head) ----------------

__global__ __launch_bounds__(32)
void attn_kernel(const _Float16* __restrict__ q16, const _Float16* __restrict__ k16,
                 const _Float16* __restrict__ vT, const unsigned* __restrict__ mask,
                 _Float16* __restrict__ attn16) {
  const int qt = blockIdx.x;
  const int hd = blockIdx.y;
  const int lane = threadIdx.x;
  const int g = lane >> 4, n = lane & 15;
  const float NEG_INF = -__builtin_inff();

  __shared__ _Float16 psh[16 * 32]; // P tile staged for C->A layout transpose
  __shared__ unsigned mrow[16];     // per-query-row 32-key mask bits

  const v16h qf = load_frag(q16, qt * 16, hd * DHEAD, DMODEL, lane);
  const _Float16* vTh = vT + (size_t)hd * DHEAD * N_NODES;

  v8f o0 = {}; // queries x dh[0..15]
  v8f o1 = {}; // queries x dh[16..31]
  float mrun[8], lrun[8];
#pragma unroll
  for (int r = 0; r < 8; ++r) { mrun[r] = NEG_INF; lrun[r] = 0.f; }

  for (int kt = 0; kt < N_NODES / 32; ++kt) {
    // scores for 32 keys: B-frag of K^T == A-layout load of K rows
    v16h kf0 = load_frag(k16, kt * 32, hd * DHEAD, DMODEL, lane);
    v16h kf1 = load_frag(k16, kt * 32 + 16, hd * DHEAD, DMODEL, lane);
    v8f s0 = wmma_f16(qf, kf0, v8f{});
    v8f s1 = wmma_f16(qf, kf1, v8f{});

    if (lane < 16)
      mrow[lane] = mask[(size_t)(qt * 16 + lane) * WORDS_PER_ROW + kt];
    __syncthreads();

#pragma unroll
    for (int r = 0; r < 8; ++r) {
      const int row = r + g * 8;
      const unsigned mb = mrow[row];
      float x0 = ((mb >> n) & 1u) ? s0[r] : NEG_INF;
      float x1 = ((mb >> (n + 16)) & 1u) ? s1[r] : NEG_INF;
      float mx = fmaxf(x0, x1);
#pragma unroll
      for (int off = 1; off < 16; off <<= 1)
        mx = fmaxf(mx, __shfl_xor(mx, off, 16));
      const float mold = mrun[r];
      const float mnew = fmaxf(mold, mx);
      float corr, p0, p1;
      if (mnew == NEG_INF) { // tile + history fully masked
        corr = 1.f; p0 = 0.f; p1 = 0.f;
      } else {
        corr = (mold == NEG_INF) ? 0.f : __expf(mold - mnew);
        p0 = (x0 == NEG_INF) ? 0.f : __expf(x0 - mnew);
        p1 = (x1 == NEG_INF) ? 0.f : __expf(x1 - mnew);
      }
      mrun[r] = mnew;
      float rs = p0 + p1;
#pragma unroll
      for (int off = 1; off < 16; off <<= 1) rs += __shfl_xor(rs, off, 16);
      lrun[r] = lrun[r] * corr + rs;
      o0[r] *= corr;
      o1[r] *= corr;
      psh[row * 32 + n] = (_Float16)p0;
      psh[row * 32 + 16 + n] = (_Float16)p1;
    }
    __syncthreads();

    v16h pf = load_frag(psh, 0, 0, 32, lane);               // P as A-frag
    v16h vf0 = load_frag(vTh, 0, kt * 32, N_NODES, lane);   // V cols 0..15
    v16h vf1 = load_frag(vTh, 16, kt * 32, N_NODES, lane);  // V cols 16..31
    o0 = wmma_f16(pf, vf0, o0);
    o1 = wmma_f16(pf, vf1, o1);
    __syncthreads();
  }

#pragma unroll
  for (int r = 0; r < 8; ++r) {
    const int row = qt * 16 + r + g * 8;
    const float inv = 1.0f / lrun[r]; // self-loops guarantee lrun > 0
    attn16[(size_t)row * DMODEL + hd * DHEAD + n] = (_Float16)(o0[r] * inv);
    attn16[(size_t)row * DMODEL + hd * DHEAD + 16 + n] = (_Float16)(o1[r] * inv);
  }
}

// ---------------- output projection + residual + LayerNorm ----------------

__global__ __launch_bounds__(32)
void out_ln_kernel(const _Float16* __restrict__ attn16,
                   const _Float16* __restrict__ WoT,
                   const float* __restrict__ bo, const float* __restrict__ hin,
                   const float* __restrict__ gamma, const float* __restrict__ beta,
                   float* __restrict__ out) {
  const int bt = blockIdx.x; // node row tile
  const int lane = threadIdx.x;
  const int g = lane >> 4, n = lane & 15;

  v16h a[4];
#pragma unroll
  for (int kk = 0; kk < 4; ++kk)
    a[kk] = load_frag(attn16, bt * 16, kk * 32, DMODEL, lane);

  float acc[8][8]; // [col tile][r]
#pragma unroll
  for (int t = 0; t < 8; ++t) {
    v8f c = {};
#pragma unroll
    for (int kk = 0; kk < 4; ++kk) {
      v16h b = load_frag(WoT, t * 16, kk * 32, DMODEL, lane);
      c = wmma_f16(a[kk], b, c);
    }
    const int col = t * 16 + n;
    const float bcol = bo[col];
#pragma unroll
    for (int r = 0; r < 8; ++r) {
      const int row = bt * 16 + r + g * 8;
      acc[t][r] = c[r] + bcol + hin[(size_t)row * DMODEL + col]; // + residual
    }
  }
#pragma unroll
  for (int r = 0; r < 8; ++r) {
    float s = 0.f, sq = 0.f;
#pragma unroll
    for (int t = 0; t < 8; ++t) { s += acc[t][r]; sq += acc[t][r] * acc[t][r]; }
#pragma unroll
    for (int off = 1; off < 16; off <<= 1) {
      s += __shfl_xor(s, off, 16);
      sq += __shfl_xor(sq, off, 16);
    }
    const float mean = s * (1.0f / 128.0f);
    const float var = sq * (1.0f / 128.0f) - mean * mean;
    const float rstd = rsqrtf(var + 1e-5f);
    const int row = bt * 16 + r + g * 8;
#pragma unroll
    for (int t = 0; t < 8; ++t) {
      const int col = t * 16 + n;
      out[(size_t)row * DMODEL + col] =
          (acc[t][r] - mean) * rstd * gamma[col] + beta[col];
    }
  }
}

// ---------------- host ----------------

extern "C" void kernel_launch(void* const* d_in, const int* in_sizes, int n_in,
                              void* d_out, int out_size, void* d_ws, size_t ws_size,
                              hipStream_t stream) {
  const float* h = (const float*)d_in[0];
  // d_in[1]=e_feat, d_in[12]=We, d_in[13]=be: the per-(node,head) edge bias is
  // added uniformly over the softmax key axis -> exactly cancelled by the
  // max-subtracting softmax. Skipped (bitwise-equivalent result).
  const int* src = (const int*)d_in[2];
  const int* dst = (const int*)d_in[3];
  const float* Wq = (const float*)d_in[4];  const float* bq = (const float*)d_in[5];
  const float* Wk = (const float*)d_in[6];  const float* bk = (const float*)d_in[7];
  const float* Wv = (const float*)d_in[8];  const float* bv = (const float*)d_in[9];
  const float* Wo = (const float*)d_in[10]; const float* bo = (const float*)d_in[11];
  const float* gamma = (const float*)d_in[14];
  const float* beta = (const float*)d_in[15];
  float* out = (float*)d_out;
  const int E = in_sizes[2];

  // workspace carve-up (~12.7 MB total)
  char* ws = (char*)d_ws;
  size_t off = 0;
  auto carve = [&](size_t bytes) -> char* {
    char* p = ws + off;
    off = (off + bytes + 255) & ~(size_t)255;
    return p;
  };
  unsigned* maskbits = (unsigned*)carve((size_t)N_NODES * WORDS_PER_ROW * 4);
  _Float16* h16 = (_Float16*)carve((size_t)N_NODES * DMODEL * 2);
  _Float16* q16 = (_Float16*)carve((size_t)N_NODES * DMODEL * 2);
  _Float16* k16 = (_Float16*)carve((size_t)N_NODES * DMODEL * 2);
  _Float16* vT = (_Float16*)carve((size_t)N_NODES * DMODEL * 2);
  _Float16* attn16 = (_Float16*)carve((size_t)N_NODES * DMODEL * 2);
  _Float16* WqT = (_Float16*)carve((size_t)DMODEL * DMODEL * 2);
  _Float16* WkT = (_Float16*)carve((size_t)DMODEL * DMODEL * 2);
  _Float16* WvT = (_Float16*)carve((size_t)DMODEL * DMODEL * 2);
  _Float16* WoT = (_Float16*)carve((size_t)DMODEL * DMODEL * 2);

  const int nwords = N_NODES * WORDS_PER_ROW;
  zero_mask_kernel<<<1024, 256, 0, stream>>>(maskbits, nwords);
  build_mask_kernel<<<(E + 255) / 256, 256, 0, stream>>>(src, dst, maskbits, E);

  cvt_f16_kernel<<<1024, 256, 0, stream>>>(h, h16, N_NODES * DMODEL);
  transpose_w_kernel<<<64, 256, 0, stream>>>(Wq, WqT);
  transpose_w_kernel<<<64, 256, 0, stream>>>(Wk, WkT);
  transpose_w_kernel<<<64, 256, 0, stream>>>(Wv, WvT);
  transpose_w_kernel<<<64, 256, 0, stream>>>(Wo, WoT);

  qkv_kernel<<<dim3(DMODEL / 16, N_NODES / 16, 3), 32, 0, stream>>>(
      h16, WqT, WkT, WvT, bq, bk, bv, q16, k16, vT);

  attn_kernel<<<dim3(N_NODES / 16, NHEADS), 32, 0, stream>>>(q16, k16, vT,
                                                             maskbits, attn16);

  out_ln_kernel<<<N_NODES / 16, 32, 0, stream>>>(attn16, WoT, bo, h, gamma,
                                                 beta, out);
}
